// WindowAttention_17265768530054
// MI455X (gfx1250) — compile-verified, matrix operand order checked by reading, not run on hardware
//
#include <hip/hip_runtime.h>
#include <cstdint>

// ---- problem constants (from reference) ----
#define NTOK        64            // window 8x8
#define NHEADS      8
#define HDIM        32
#define ROW_QKV     768           // 3*DIM floats per token in qkv
#define ROW_DIM     256           // DIM floats per token
#define LOGIT_CLAMP 4.6051701859880914f   // log(1/0.01)
#define QK_SCALE    0.17677669529663687f  // 32^-0.5

typedef __attribute__((ext_vector_type(16))) __bf16        v16bf;
typedef __attribute__((ext_vector_type(8)))  float         v8f;
typedef __attribute__((ext_vector_type(4)))  float         v4f;
typedef __attribute__((ext_vector_type(4)))  unsigned int  v4u;

union BF16X16 { v16bf v; unsigned int w[8]; unsigned short u[16]; };
union V8FU    { v8f v; v4f h[2]; };

// Pack two f32 -> one dword of two bf16 (round-half-up via +0x8000, then
// byte-select the two high halves with a single v_perm_b32).
__device__ __forceinline__ unsigned int pack2bf(float a, float b) {
  unsigned int ua = __builtin_bit_cast(unsigned int, a) + 0x8000u;
  unsigned int ub = __builtin_bit_cast(unsigned int, b) + 0x8000u;
  // S0=ub (bytes 4..7), S1=ua (bytes 0..3): out = { ub[3],ub[2], ua[3],ua[2] }
  return __builtin_amdgcn_perm(ub, ua, 0x07060302u);
}

__device__ __forceinline__ unsigned short f2bf_bits(float f) {
  unsigned int x = __builtin_bit_cast(unsigned int, f) + 0x8000u;
  return (unsigned short)(x >> 16);
}

// A/B fragment from a contiguous f32 row (row-major [*,32] matrix), times `scale`.
// 16-bit A layout: elems 0..7 <- row[hi*8+0..7], elems 8..15 <- row[16+hi*8+0..7]
__device__ __forceinline__ v16bf load_frag_row(const float* __restrict__ rowp, int hi,
                                               float scale) {
  const v4f* p0 = (const v4f*)(rowp + hi * 8);
  const v4f* p1 = (const v4f*)(rowp + 16 + hi * 8);
  v4f f0 = __builtin_nontemporal_load(p0);
  v4f f1 = __builtin_nontemporal_load(p0 + 1);
  v4f f2 = __builtin_nontemporal_load(p1);
  v4f f3 = __builtin_nontemporal_load(p1 + 1);
  f0 *= scale; f1 *= scale; f2 *= scale; f3 *= scale;   // folded away when scale==1.0f
  BF16X16 a;
  a.w[0] = pack2bf(f0.x, f0.y);  a.w[1] = pack2bf(f0.z, f0.w);
  a.w[2] = pack2bf(f1.x, f1.y);  a.w[3] = pack2bf(f1.z, f1.w);
  a.w[4] = pack2bf(f2.x, f2.y);  a.w[5] = pack2bf(f2.z, f2.w);
  a.w[6] = pack2bf(f3.x, f3.y);  a.w[7] = pack2bf(f3.z, f3.w);
  return a.v;
}

// B fragment where the K dimension walks rows of a [64,*] f32 matrix (stride 768),
// column fixed per lane (coalesced across lanes 0..15 per element slot).
__device__ __forceinline__ v16bf load_frag_strided(const float* __restrict__ base, int hi) {
  float f[16];
#pragma unroll
  for (int e = 0; e < 8; ++e) f[e]     = __builtin_nontemporal_load(base + (hi * 8 + e) * ROW_QKV);
#pragma unroll
  for (int e = 0; e < 8; ++e) f[8 + e] = __builtin_nontemporal_load(base + (16 + hi * 8 + e) * ROW_QKV);
  BF16X16 a;
#pragma unroll
  for (int e = 0; e < 8; ++e) a.w[e] = pack2bf(f[2 * e], f[2 * e + 1]);
  return a.v;
}

// Bias tile in WMMA C layout: 8 consecutive floats per lane -> v8f accumulator.
__device__ __forceinline__ v8f load_bias_c(const float* __restrict__ p) {
  V8FU u;
  u.h[0] = *(const v4f*)p;
  u.h[1] = *(const v4f*)(p + 4);
  return u.v;
}

// LDS P matrix: row-major [64][64] bf16, 8-element (16B) chunks XOR-swizzled by row.
__device__ __forceinline__ void store_P(unsigned short* Pw, int row, int col, unsigned short bits) {
  const int swz = (row ^ (row >> 3)) & 7;
  const int c = ((col >> 3) ^ swz);
  Pw[row * 64 + c * 8 + (col & 7)] = bits;
}

__device__ __forceinline__ v16bf load_frag_P(const unsigned short* __restrict__ Pw,
                                             int row, int kc, int hi) {
  const int swz = (row ^ (row >> 3)) & 7;
  const int c0 = (kc * 4 + hi) ^ swz;        // cols kc*32 + hi*8 + 0..7
  const int c1 = (kc * 4 + 2 + hi) ^ swz;    // cols kc*32 + 16 + hi*8 + 0..7
  v4u t0 = *(const v4u*)(Pw + row * 64 + c0 * 8);   // ds_read_b128
  v4u t1 = *(const v4u*)(Pw + row * 64 + c1 * 8);
  BF16X16 a;
  a.w[0] = t0.x; a.w[1] = t0.y; a.w[2] = t0.z; a.w[3] = t0.w;
  a.w[4] = t1.x; a.w[5] = t1.y; a.w[6] = t1.z; a.w[7] = t1.w;
  return a.v;
}

// ---------------- bias MLP (batch independent) ----------------
// Written directly in WMMA C-layout lane order:
//   bias[(((h*4 + tr)*4 + tc)*32 + lane)*8 + r]
// so an attention lane's 8 per-row values are two b128 loads (and a valid v8f C tile).
__global__ void bias_kernel(const float* __restrict__ w1, const float* __restrict__ b1,
                            const float* __restrict__ w2, const float* __restrict__ b2,
                            float* __restrict__ bias) {
  int idx = blockIdx.x * blockDim.x + threadIdx.x;
  if (idx >= NTOK * NTOK) return;
  int n = idx >> 6, m = idx & 63;             // query row n, key col m
  float dy = (float)((n >> 3) - (m >> 3));
  float dx = (float)((n & 7) - (m & 7));
  float r0 = copysignf(log1pf(fabsf(dy)), dy);
  float r1 = copysignf(log1pf(fabsf(dx)), dx);
  float acc[NHEADS] = {0.f, 0.f, 0.f, 0.f, 0.f, 0.f, 0.f, 0.f};
  for (int j = 0; j < 256; ++j) {
    float hv = fmaxf(0.f, r0 * w1[j] + r1 * w1[256 + j] + b1[j]);
#pragma unroll
    for (int hh = 0; hh < NHEADS; ++hh) acc[hh] += hv * w2[j * NHEADS + hh];
  }
  const int tr = n >> 4, r = n & 7, hi2 = (n >> 3) & 1;
  const int tc = m >> 4, l16 = m & 15;
  const int lane = hi2 * 16 + l16;
#pragma unroll
  for (int hh = 0; hh < NHEADS; ++hh)
    bias[(size_t)(((hh * 4 + tr) * 4 + tc) * 32 + lane) * 8 + r] = acc[hh] + b2[hh];
}

// ---------------- fused window attention: 1 block/window, 1 wave/head ----------------
__global__ __launch_bounds__(256) void attn_kernel(const float* __restrict__ qkv,
                                                   const float* __restrict__ convo,
                                                   const float* __restrict__ logit_scale,
                                                   const float* __restrict__ bias,
                                                   float* __restrict__ out) {
  __shared__ unsigned short P[NHEADS * NTOK * NTOK];  // 64 KB, per-wave private slabs
  const int b    = blockIdx.x;
  const int w    = threadIdx.x >> 5;     // wave == head
  const int lane = threadIdx.x & 31;
  const int l16  = lane & 15;
  const int hi   = lane >> 4;
  const int h    = w;
  unsigned short* Pw = P + w * (NTOK * NTOK);

  const float ls = __expf(fminf(logit_scale[0], LOGIT_CLAMP));
  const float sc = QK_SCALE * ls;        // folded into Q at bf16-conversion time
  const size_t rowbase = (size_t)b * NTOK;

  // K fragments: B operand of S = Q*K^T (per tile of 16 key positions)
  v16bf bk[4];
#pragma unroll
  for (int tc = 0; tc < 4; ++tc) {
    const float* kr = qkv + (rowbase + 16 * tc + l16) * ROW_QKV + 256 + h * HDIM;
    bk[tc] = load_frag_row(kr, hi, 1.0f);
  }

  // ---- phase 1: S = (Q*sc)*K^T + bias (bias via WMMA C operand),
  //      row softmax (un-normalized), P=exp(...) -> bf16 in LDS,
  //      per-row 1/sum kept in registers for phase 2. ----
  float invs[4][8];
#pragma unroll
  for (int tr = 0; tr < 4; ++tr) {
    const float* qr = convo + (rowbase + 16 * tr + l16) * ROW_DIM + h * HDIM;
    v16bf aq = load_frag_row(qr, hi, sc);
    v8f s[4];
#pragma unroll
    for (int tc = 0; tc < 4; ++tc) {
      v8f cbias = load_bias_c(bias + (size_t)(((h * 4 + tr) * 4 + tc) * 32 + lane) * 8);
      s[tc] = __builtin_amdgcn_wmma_f32_16x16x32_bf16(false, aq, false, bk[tc],
                                                      (short)0, cbias, false, false);
    }
#pragma unroll
    for (int r = 0; r < 8; ++r) {
      const int row = 16 * tr + hi * 8 + r;   // query index; 64 logits on 16 lanes x 4 tiles
      float mx = fmaxf(fmaxf(s[0][r], s[1][r]), fmaxf(s[2][r], s[3][r]));
#pragma unroll
      for (int off = 1; off < 16; off <<= 1) mx = fmaxf(mx, __shfl_xor(mx, off, 16));
      float e0 = __expf(s[0][r] - mx), e1 = __expf(s[1][r] - mx);
      float e2 = __expf(s[2][r] - mx), e3 = __expf(s[3][r] - mx);
      float sum = (e0 + e1) + (e2 + e3);
#pragma unroll
      for (int off = 1; off < 16; off <<= 1) sum += __shfl_xor(sum, off, 16);
      invs[tr][r] = 1.0f / sum;               // deferred normalization
      store_P(Pw, row, 0 * 16 + l16, f2bf_bits(e0));
      store_P(Pw, row, 1 * 16 + l16, f2bf_bits(e1));
      store_P(Pw, row, 2 * 16 + l16, f2bf_bits(e2));
      store_P(Pw, row, 3 * 16 + l16, f2bf_bits(e3));
    }
  }

  // V fragments: B operand of O = P*V (loaded here to keep live ranges short)
  v16bf bv[2][2];
#pragma unroll
  for (int kc = 0; kc < 2; ++kc)
#pragma unroll
    for (int dc = 0; dc < 2; ++dc) {
      const float* vb = qkv + (rowbase + kc * 32) * ROW_QKV + 512 + h * HDIM + dc * 16 + l16;
      bv[kc][dc] = load_frag_strided(vb, hi);
    }

  const v8f zero = {0.f, 0.f, 0.f, 0.f, 0.f, 0.f, 0.f, 0.f};

  // ---- phase 2: O = (E @ V) * inv_rowsum (K=64 via two chained K=32 WMMAs) ----
#pragma unroll
  for (int tr = 0; tr < 4; ++tr) {
    const int arow = 16 * tr + l16;
    v16bf pa0 = load_frag_P(Pw, arow, 0, hi);
    v16bf pa1 = load_frag_P(Pw, arow, 1, hi);
#pragma unroll
    for (int dc = 0; dc < 2; ++dc) {
      v8f c = __builtin_amdgcn_wmma_f32_16x16x32_bf16(false, pa0, false, bv[0][dc],
                                                      (short)0, zero, false, false);
      c = __builtin_amdgcn_wmma_f32_16x16x32_bf16(false, pa1, false, bv[1][dc],
                                                  (short)0, c, false, false);
#pragma unroll
      for (int r = 0; r < 8; ++r) {
        const int row = 16 * tr + hi * 8 + r;   // same (tr,r)<->row map as phase 1
        __builtin_nontemporal_store(
            c[r] * invs[tr][r],
            out + (rowbase + row) * ROW_DIM + h * HDIM + dc * 16 + l16);
      }
    }
  }
}

extern "C" void kernel_launch(void* const* d_in, const int* in_sizes, int n_in,
                              void* d_out, int out_size, void* d_ws, size_t ws_size,
                              hipStream_t stream) {
  (void)n_in; (void)out_size; (void)ws_size;
  const float* qkv         = (const float*)d_in[0];
  const float* convo       = (const float*)d_in[1];
  const float* logit_scale = (const float*)d_in[2];
  const float* w1          = (const float*)d_in[3];
  const float* b1          = (const float*)d_in[4];
  const float* w2          = (const float*)d_in[5];
  const float* b2          = (const float*)d_in[6];
  float* bias = (float*)d_ws;                       // 8*64*64 f32 = 128 KB scratch
  const int B = in_sizes[0] / (NTOK * ROW_QKV);     // 4096

  bias_kernel<<<(NTOK * NTOK + 255) / 256, 256, 0, stream>>>(w1, b1, w2, b2, bias);
  attn_kernel<<<B, 256, 0, stream>>>(qkv, convo, logit_scale, bias, (float*)d_out);
}
